// NewAttention_55344948576827
// MI455X (gfx1250) — matrix operand order repaired and checked by reading, not compiled, not run on hardware
//
#include <hip/hip_runtime.h>
#include <hip/hip_bf16.h>

// ---- CDNA5 WMMA vector types (wave32) ----
typedef __attribute__((ext_vector_type(16))) __bf16 v16bf;
typedef __attribute__((ext_vector_type(8)))  __bf16 v8bf;
typedef __attribute__((ext_vector_type(8)))  float  v8f;
typedef __attribute__((ext_vector_type(2)))  float  v2f;
// ---- TDM descriptor vector types ----
typedef __attribute__((ext_vector_type(4))) unsigned int ui32x4;
typedef __attribute__((ext_vector_type(8))) int          i32x8;
typedef __attribute__((ext_vector_type(4))) int          i32x4;

#define SEQ    4096
#define EMB    512
#define BATCH  4
#define ROWS   (BATCH * SEQ)       // 16384
#define RADIUS 10                  // exp(-0.5*10^2) ~ 2e-22 -> band fully covered
#define TAPS   (2 * RADIUS + 1)

// GEMM tiling
#define BN       64                // N cols per block (LDS-staged B slab)
#define BSTRIDE  528               // padded LDS row stride (elems): 512 + 16
// Conv tiling
#define TS   64                    // seq positions per block
#define CH   128                   // channels per block

// -------------------------------------------------------------------------
// Kernel 1: Wf[n,k] = sum_e Wo[n,e] * Wi[e,k]   (512x512, fp32 WMMA)
// -------------------------------------------------------------------------
__global__ __launch_bounds__(256) void fuse_weights_f32wmma(
    const float* __restrict__ Wi, const float* __restrict__ Wo,
    __bf16* __restrict__ Wf)
{
    const int lane = threadIdx.x & 31;
    const int wv   = blockIdx.x * 8 + (threadIdx.x >> 5);  // 0..1023 waves
    const int nt   = wv >> 5;
    const int kt   = wv & 31;
    const int g    = lane >> 4;
    const int r    = lane & 15;

    v8f c = {};
    const float* Arow = Wo + (size_t)(nt * 16 + r) * EMB;
    const int    kcol = kt * 16 + r;

    for (int e = 0; e < EMB; e += 4) {
        v2f a = *(const v2f*)(Arow + e + 2 * g);        // A[r, e+2g], A[r, e+2g+1]
        v2f b;
        b[0] = Wi[(size_t)(e + 2 * g)     * EMB + kcol];
        b[1] = Wi[(size_t)(e + 2 * g + 1) * EMB + kcol];
        c = __builtin_amdgcn_wmma_f32_16x16x4_f32(
                false, a, false, b, (short)0, c, false, false);
    }
#pragma unroll
    for (int j = 0; j < 8; ++j) {
        const int n = nt * 16 + g * 8 + j;
        Wf[(size_t)n * EMB + kcol] = (__bf16)c[j];
    }
}

// -------------------------------------------------------------------------
// Kernel 2: Gaussian band conv along S, LDS-tiled (each X element read ~1.3x).
// Block: TS=64 seq positions x CH=128 channels; halo RADIUS each side.
// -------------------------------------------------------------------------
__global__ __launch_bounds__(256) void gauss_conv_s(
    const float* __restrict__ X, __bf16* __restrict__ Y)
{
    __shared__ float tile[(TS + 2 * RADIUS) * CH];       // 84*128*4 = 43 KB

    const int s0 = blockIdx.x * TS;
    const int c0 = blockIdx.y * CH;
    const int b  = blockIdx.z;
    const float* Xb = X + (size_t)b * SEQ * EMB + c0;

    for (int i = threadIdx.x; i < (TS + 2 * RADIUS) * CH; i += 256) {
        const int row = i >> 7;                          // CH == 128
        const int col = i & (CH - 1);
        const int ss  = s0 - RADIUS + row;
        tile[i] = (ss >= 0 && ss < SEQ) ? Xb[(size_t)ss * EMB + col] : 0.f;
    }
    __syncthreads();

    float w[TAPS];
#pragma unroll
    for (int j = 0; j < TAPS; ++j) {
        const int dx = j - RADIUS;
        w[j] = 0.3989422804014327f * __expf(-0.5f * (float)(dx * dx));
    }

    const int col = threadIdx.x & (CH - 1);
    const int rg  = threadIdx.x >> 7;                    // 0/1
    for (int rr = 0; rr < TS / 2; ++rr) {
        const int sl = rr * 2 + rg;
        float acc = 0.f;
#pragma unroll
        for (int j = 0; j < TAPS; ++j)
            acc += w[j] * tile[(sl + j) * CH + col];
        Y[((size_t)b * SEQ + s0 + sl) * EMB + c0 + col] = (__bf16)acc;
    }
}

// -------------------------------------------------------------------------
// Kernel 3: out[m,n] = sum_k Y[m,k] * Wf[n,k]   (16384x512x512, bf16 WMMA)
// Block = 8 waves = 128 M-rows x 64 N-cols. B slab (64x512 bf16) staged once
// per block into LDS via the Tensor Data Mover, with D# LDS-padding so the
// padded row stride (528 elems) gives conflict-free ds_load_b128 fragments.
// -------------------------------------------------------------------------
__global__ __launch_bounds__(256) void gemm_bf16_wmma(
    const __bf16* __restrict__ Y, const __bf16* __restrict__ Wf,
    float* __restrict__ out)
{
    __shared__ __bf16 Bs[BN * BSTRIDE];                  // 64*528*2 = 66 KB

    const int lane  = threadIdx.x & 31;
    const int wv    = threadIdx.x >> 5;
    const int g     = lane >> 4;
    const int r     = lane & 15;
    const int nBase = blockIdx.y * BN;
    const int mBase = blockIdx.x * 128 + wv * 16;

#if __has_builtin(__builtin_amdgcn_tensor_load_to_lds)
    // ---- TDM: 2D tile load Wf[nBase..nBase+63][0..511] -> padded LDS ----
    if (threadIdx.x < 32) {
        const unsigned ldsOff = (unsigned)(unsigned long long)Bs; // addr[31:0] = LDS offset
        const unsigned long long ga =
            (unsigned long long)(Wf + (size_t)nBase * EMB);
        ui32x4 g0;
        g0[0] = 1u;                                   // count=1, user descriptor
        g0[1] = ldsOff;                               // lds_addr [63:32]
        g0[2] = (unsigned)(ga & 0xffffffffu);         // global_addr lo
        g0[3] = (unsigned)((ga >> 32) & 0x01ffffffu)  // global_addr hi (57-bit)
              | 0x80000000u;                          // type=2 ("image") [127:126]
        i32x8 g1;
        g1[0] = 0x00010000                            // data_size=1 (2 bytes)
              | (1 << 20)                             // pad_enable
              | (7 << 22)                             // pad_interval: 256 dwords (1 row)
              | (7 << 25);                            // pad_amount: 8 dwords (16 elems)
        g1[1] = (int)(512u << 16);                    // tensor_dim0 = 512 (lo16)
        g1[2] = (int)(512u << 16);                    // dim0 hi=0 | tensor_dim1 = 512 (lo16)
        g1[3] = (int)(512u << 16);                    // dim1 hi=0 | tile_dim0 = 512
        g1[4] = BN;                                   // tile_dim1 = 64 | tile_dim2 = 0
        g1[5] = 512;                                  // tensor_dim0_stride lo32
        g1[6] = 0;                                    // stride0 hi | stride1 lo
        g1[7] = 0;                                    // stride1 hi
        i32x4 z4 = {0, 0, 0, 0};
#if __clang_major__ >= 23
        i32x8 z8 = {0, 0, 0, 0, 0, 0, 0, 0};
        __builtin_amdgcn_tensor_load_to_lds(g0, g1, z4, z4, z8, 0);
#else
        __builtin_amdgcn_tensor_load_to_lds(g0, g1, z4, z4, 0);
#endif
        __builtin_amdgcn_s_wait_tensorcnt(0);
    }
    __syncthreads();
#else
    // ---- fallback: cooperative copy into the same padded layout ----
    for (int i = threadIdx.x; i < BN * (EMB / 8); i += 256) {
        const int row = i >> 6;                          // EMB/8 == 64 chunks/row
        const int ch  = i & 63;
        *(v8bf*)(Bs + (size_t)row * BSTRIDE + ch * 8) =
            *(const v8bf*)(Wf + (size_t)(nBase + row) * EMB + ch * 8);
    }
    __syncthreads();
#endif

    v8f c0 = {}, c1 = {}, c2 = {}, c3 = {};
    const __bf16* Arow  = Y + (size_t)(mBase + r) * EMB;
    const __bf16* BsRow = Bs + (size_t)r * BSTRIDE;      // n_local = r

    for (int kk = 0; kk < EMB; kk += 32) {
        // A 16x32 bf16: elems 0..15 <- K=kk+g*16 .. +15 contiguous
        v16bf a = *(const v16bf*)(Arow + kk + g * 16);
        // B 32x16 bf16 from LDS: lane = col n; K = kk+g*16 .. +15 contiguous
        const __bf16* bp = BsRow + kk + g * 16;
        v16bf b0 = *(const v16bf*)(bp);
        v16bf b1 = *(const v16bf*)(bp + 16 * BSTRIDE);
        v16bf b2 = *(const v16bf*)(bp + 32 * BSTRIDE);
        v16bf b3 = *(const v16bf*)(bp + 48 * BSTRIDE);
        c0 = __builtin_amdgcn_wmma_f32_16x16x32_bf16(false, a, false, b0, (short)0, c0, false, false);
        c1 = __builtin_amdgcn_wmma_f32_16x16x32_bf16(false, a, false, b1, (short)0, c1, false, false);
        c2 = __builtin_amdgcn_wmma_f32_16x16x32_bf16(false, a, false, b2, (short)0, c2, false, false);
        c3 = __builtin_amdgcn_wmma_f32_16x16x32_bf16(false, a, false, b3, (short)0, c3, false, false);
    }

    // D f32 16x16: VGPR j -> (M = g*8+j, N = r); tiles at n offsets 0/16/32/48
#pragma unroll
    for (int j = 0; j < 8; ++j) {
        float* o = out + (size_t)(mBase + g * 8 + j) * EMB + nBase + r;
        o[0]  = c0[j];
        o[16] = c1[j];
        o[32] = c2[j];
        o[48] = c3[j];
    }
}

extern "C" void kernel_launch(void* const* d_in, const int* in_sizes, int n_in,
                              void* d_out, int out_size, void* d_ws, size_t ws_size,
                              hipStream_t stream) {
    const float* X  = (const float*)d_in[0];   // [4,4096,512]
    const float* Wi = (const float*)d_in[1];   // [512,512] input_weights
    const float* Wo = (const float*)d_in[2];   // [512,512] output_weight
    float* out = (float*)d_out;                // [4,4096,512]

    __bf16* Ybf  = (__bf16*)d_ws;                                   // 16 MB
    __bf16* Wfbf = (__bf16*)((char*)d_ws + (size_t)ROWS * EMB * 2); // 512 KB

    // Wf = Wo @ Wi via fp32 WMMA
    fuse_weights_f32wmma<<<128, 256, 0, stream>>>(Wi, Wo, Wfbf);
    // Gaussian positional "attention" as LDS-tiled banded conv along S
    gauss_conv_s<<<dim3(SEQ / TS, EMB / CH, BATCH), 256, 0, stream>>>(X, Ybf);
    // out = conv(X) @ Wf^T via bf16 WMMA, TDM-staged B slab in LDS
    gemm_bf16_wmma<<<dim3(ROWS / 128, EMB / BN), 256, 0, stream>>>(Ybf, Wfbf, out);
}
